// myFunction_39883066311001
// MI455X (gfx1250) — matrix operand order repaired and checked by reading, not compiled, not run on hardware
//
#include <hip/hip_runtime.h>
#include <hip/hip_bf16.h>

typedef __attribute__((ext_vector_type(16))) _Float16 v16h;
typedef __attribute__((ext_vector_type(8)))  _Float16 v8h;
typedef __attribute__((ext_vector_type(8)))  float    v8f;

#define NPART   16
#define HID     256
#define NLAYER  3          // mid layers
#define HSTRIDE 264        // 256 + 8 f16 pad (row stride, keeps 16B alignment, spreads banks)
#define SLAB_B  10240      // bytes of LDS per wave
#define LOG16FACT 30.67186010608067f

// Hardware tanh (v_tanh_f32) if the toolchain exposes it; ocml fallback otherwise.
#if __has_builtin(__builtin_amdgcn_tanhf)
#define TANHF(x) __builtin_amdgcn_tanhf(x)
#else
#define TANHF(x) tanhf(x)
#endif

// ---------------------------------------------------------------------------
// Prep: convert W_mid / W_sl to f16, pre-swizzled into the WMMA B-fragment
// lane layout: block(i,kt,nt) of 512 f16; lane l holds col n=nt*16+(l&15),
// K = kt*32 + (l>>4)*8 + (j&7) + (j>>3)*16  (j=0..15) -> one v16h (32B) per lane.
// ---------------------------------------------------------------------------
__global__ void prep_weights_k(const float* __restrict__ Wmid,
                               const float* __restrict__ Wsl,
                               _Float16* __restrict__ wsMid,
                               _Float16* __restrict__ wsSl) {
    int idx = blockIdx.x * blockDim.x + threadIdx.x;
    const int MIDTOT = NLAYER * 16 * 16 * 512;     // 393216
    if (idx < MIDTOT) {
        int j    = idx & 15;
        int lane = (idx >> 4) & 31;
        int bi   = idx >> 9;
        int nt = bi & 15, kt = (bi >> 4) & 15, i = bi >> 8;
        int k = kt * 32 + (lane >> 4) * 8 + (j & 7) + (j >> 3) * 16;    // 0..511
        int n = nt * 16 + (lane & 15);             // 0..255
        wsMid[idx] = (_Float16)Wmid[(size_t)i * 512 * 256 + (size_t)k * 256 + n];
    } else {
        int t = idx - MIDTOT;
        if (t < 8 * 512) {                          // W_sl: K=256 (8 kt), N=16
            int j    = t & 15;
            int lane = (t >> 4) & 31;
            int kt   = t >> 9;
            int k = kt * 32 + (lane >> 4) * 8 + (j & 7) + (j >> 3) * 16; // 0..255
            int n = lane & 15;                      // 0..15
            wsSl[t] = (_Float16)Wsl[k * 16 + n];
        }
    }
}

__device__ __forceinline__ v16h make_a(v8h lo, v8h hi) {
    return __builtin_shufflevector(lo, hi, 0,1,2,3,4,5,6,7,8,9,10,11,12,13,14,15);
}

// ---------------------------------------------------------------------------
// Main: one wave per batch element, 8 waves per block.
// ---------------------------------------------------------------------------
__global__ __launch_bounds__(256) void ferminet_fwd_k(
    const float* __restrict__ x0g,  const float* __restrict__ Win,
    const float* __restrict__ binp, const float* __restrict__ bmid,
    const float* __restrict__ bsl,  const float* __restrict__ widthp,
    const _Float16* __restrict__ wMid, const _Float16* __restrict__ wSl,
    float* __restrict__ out, int Btot)
{
    extern __shared__ char smem[];
    const int tid  = threadIdx.x;
    const int wave = tid >> 5;
    const int lane = tid & 31;
    const int b    = blockIdx.x * 8 + wave;

    char*     slab = smem + wave * SLAB_B;
    _Float16* hb   = (_Float16*)slab;              // 16 x HSTRIDE f16  (8448 B)
    _Float16* mrow = (_Float16*)(slab + 8448);     // 256 f16           (512 B)
    float*    sl   = (float*)(slab + 8960);        // 16 x 17 f32       (1088 B)
    int*      piv  = (int*)(slab + 10048);

    const int m16 = lane & 15;
    const int hiL = lane >> 4;         // 0: K 0-7 / 16-23, 1: K 8-15 / 24-31
    const int kb  = hiL * 8;

    // ---- load x0 row, mean and sum-of-squares via lane shuffles ----
    float xv = x0g[(size_t)b * NPART + m16];
    float s1 = xv, s2 = xv * xv;
    #pragma unroll
    for (int off = 8; off >= 1; off >>= 1) {
        s1 += __shfl_xor(s1, off, 16);
        s2 += __shfl_xor(s2, off, 16);
    }
    const float meanx = s1 * (1.0f / 16.0f);
    const float sumsq = s2;

    float xr[8];                       // xr[v] = x0[v + 8*hiL] (row for C-tile vgpr v)
    #pragma unroll
    for (int v = 0; v < 8; ++v) xr[v] = __shfl(xv, v + 8 * hiL, 32);

    // ---- input layer: h = tanh(f @ W_in + b_in), C-tile layout ----
    v8f hacc[16];
    #pragma unroll
    for (int nt = 0; nt < 16; ++nt) {
        int c = nt * 16 + m16;
        float w0 = Win[c], w1 = Win[256 + c];
        float base = meanx * w1 + binp[c];
        #pragma unroll
        for (int v = 0; v < 8; ++v) hacc[nt][v] = TANHF(xr[v] * w0 + base);
    }

    // ---- 3 residual mid layers ----
    for (int i = 0; i < NLAYER; ++i) {
        // stage old h as f16 row-major in LDS; build mean row
        #pragma unroll
        for (int nt = 0; nt < 16; ++nt) {
            float s = 0.f;
            #pragma unroll
            for (int v = 0; v < 8; ++v) {
                hb[(v + 8 * hiL) * HSTRIDE + nt * 16 + m16] = (_Float16)hacc[nt][v];
                s += hacc[nt][v];
            }
            s += __shfl_xor(s, 16, 32);                  // full column sum
            if (lane < 16) mrow[nt * 16 + lane] = (_Float16)(s * (1.0f / 16.0f));
        }
        __syncthreads();

        const _Float16* wL = wMid + (size_t)i * 16 * 16 * 512;
        #pragma unroll
        for (int nt = 0; nt < 16; ++nt) {
            float bb = bmid[i * 256 + nt * 16 + m16];
            // two independent accumulator chains -> 2-way WMMA ILP per wave
            v8f acc0 = { bb, bb, bb, bb, bb, bb, bb, bb };
            v8f acc1 = { 0.f, 0.f, 0.f, 0.f, 0.f, 0.f, 0.f, 0.f };
            #pragma unroll
            for (int kt = 0; kt < 16; kt += 2) {
                // even kt -> acc0
                {
                    const _Float16* p = (kt < 8)
                        ? hb + m16 * HSTRIDE + kt * 32 + kb
                        : mrow + (kt - 8) * 32 + kb;
                    v8h lo = *(const v8h*)p;
                    v8h hi = *(const v8h*)(p + 16);
                    v16h a  = make_a(lo, hi);
                    v16h bf = *(const v16h*)(wL + ((size_t)(kt * 16 + nt)) * 512 + lane * 16);
                    acc0 = __builtin_amdgcn_wmma_f32_16x16x32_f16(
                               false, a, false, bf, (short)0, acc0, false, false);
                }
                // odd kt -> acc1
                {
                    const int k1 = kt + 1;
                    const _Float16* p = (k1 < 8)
                        ? hb + m16 * HSTRIDE + k1 * 32 + kb
                        : mrow + (k1 - 8) * 32 + kb;
                    v8h lo = *(const v8h*)p;
                    v8h hi = *(const v8h*)(p + 16);
                    v16h a  = make_a(lo, hi);
                    v16h bf = *(const v16h*)(wL + ((size_t)(k1 * 16 + nt)) * 512 + lane * 16);
                    acc1 = __builtin_amdgcn_wmma_f32_16x16x32_f16(
                               false, a, false, bf, (short)0, acc1, false, false);
                }
            }
            #pragma unroll
            for (int v = 0; v < 8; ++v)
                hacc[nt][v] = TANHF(acc0[v] + acc1[v]) + hacc[nt][v];
        }
        __syncthreads();
    }

    // ---- slater = h @ W_sl + b_sl : one 16x16 C-tile ----
    #pragma unroll
    for (int nt = 0; nt < 16; ++nt)
        #pragma unroll
        for (int v = 0; v < 8; ++v)
            hb[(v + 8 * hiL) * HSTRIDE + nt * 16 + m16] = (_Float16)hacc[nt][v];
    __syncthreads();

    {
        float bb = bsl[m16];
        v8f acc0 = { bb, bb, bb, bb, bb, bb, bb, bb };
        v8f acc1 = { 0.f, 0.f, 0.f, 0.f, 0.f, 0.f, 0.f, 0.f };
        #pragma unroll
        for (int kt = 0; kt < 8; kt += 2) {
            {
                const _Float16* p = hb + m16 * HSTRIDE + kt * 32 + kb;
                v8h lo = *(const v8h*)p;
                v8h hi = *(const v8h*)(p + 16);
                v16h a  = make_a(lo, hi);
                v16h bf = *(const v16h*)(wSl + (size_t)kt * 512 + lane * 16);
                acc0 = __builtin_amdgcn_wmma_f32_16x16x32_f16(
                           false, a, false, bf, (short)0, acc0, false, false);
            }
            {
                const int k1 = kt + 1;
                const _Float16* p = hb + m16 * HSTRIDE + k1 * 32 + kb;
                v8h lo = *(const v8h*)p;
                v8h hi = *(const v8h*)(p + 16);
                v16h a  = make_a(lo, hi);
                v16h bf = *(const v16h*)(wSl + (size_t)k1 * 512 + lane * 16);
                acc1 = __builtin_amdgcn_wmma_f32_16x16x32_f16(
                           false, a, false, bf, (short)0, acc1, false, false);
            }
        }
        #pragma unroll
        for (int v = 0; v < 8; ++v)
            sl[(v + 8 * hiL) * 17 + m16] = acc0[v] + acc1[v];
    }
    __syncthreads();

    // ---- slogdet via partial-pivot LU (rows parallel across lanes 0-15) ----
    int nswap = 0;
    for (int k = 0; k < 16; ++k) {
        if (lane == 0) {
            int p = k; float best = fabsf(sl[k * 17 + k]);
            for (int r = k + 1; r < 16; ++r) {
                float a = fabsf(sl[r * 17 + k]);
                if (a > best) { best = a; p = r; }
            }
            *piv = p;
        }
        __syncthreads();
        int p = *piv;
        if (p != k && lane < 16) {
            float t = sl[k * 17 + lane];
            sl[k * 17 + lane] = sl[p * 17 + lane];
            sl[p * 17 + lane] = t;
        }
        if (lane == 0 && p != k) nswap++;
        __syncthreads();
        float pv = sl[k * 17 + k];
        if (lane > k && lane < 16) {
            float mlt = sl[lane * 17 + k] / pv;
            sl[lane * 17 + k] = mlt;
            for (int j = k + 1; j < 16; ++j)
                sl[lane * 17 + j] -= mlt * sl[k * 17 + j];
        }
        __syncthreads();
    }

    if (lane == 0) {
        float sgn = (nswap & 1) ? -1.f : 1.f;
        float la  = 0.f;
        for (int k = 0; k < 16; ++k) {
            float d = sl[k * 17 + k];
            if (d < 0.f) sgn = -sgn;
            if (d == 0.f) sgn = 0.f;
            la += logf(fabsf(d));
        }
        float w = widthp[0];
        out[b]        = sgn;
        out[Btot + b] = la - 0.5f * LOG16FACT - w * sumsq;
    }
}

extern "C" void kernel_launch(void* const* d_in, const int* in_sizes, int n_in,
                              void* d_out, int out_size, void* d_ws, size_t ws_size,
                              hipStream_t stream) {
    const float* x0    = (const float*)d_in[0];
    const float* Win   = (const float*)d_in[1];
    const float* binp  = (const float*)d_in[2];
    const float* Wmid  = (const float*)d_in[3];
    const float* bmid  = (const float*)d_in[4];
    const float* Wsl   = (const float*)d_in[5];
    const float* bsl   = (const float*)d_in[6];
    const float* width = (const float*)d_in[7];
    float* out = (float*)d_out;

    int Bt = in_sizes[0] / NPART;          // 16384

    _Float16* wsMid = (_Float16*)d_ws;
    _Float16* wsSl  = wsMid + (size_t)NLAYER * 16 * 16 * 512;

    const int prepTot = NLAYER * 16 * 16 * 512 + 8 * 512;
    prep_weights_k<<<(prepTot + 255) / 256, 256, 0, stream>>>(Wmid, Wsl, wsMid, wsSl);

    ferminet_fwd_k<<<Bt / 8, 256, 8 * SLAB_B, stream>>>(
        x0, Win, binp, bmid, bsl, width, wsMid, wsSl, out, Bt);
}
